// ProbSparseAttention_58591943852558
// MI455X (gfx1250) — matrix operand order repaired and checked by reading, not compiled, not run on hardware
//
#include <hip/hip_runtime.h>
#include <hip/hip_bf16.h>
#include <math.h>

typedef __attribute__((ext_vector_type(16))) __bf16 v16bf;
typedef __attribute__((ext_vector_type(8)))  float  v8f;

#define D_MODEL   1024
#define NHEAD     16
#define HEAD_DIM  64
#define B_SZ      4
#define SEQ       2048
#define TOP_U     38            // int(5 * ln(2048))
#define SCALE     0.125f        // 1/sqrt(64)
#define LOG_SEQ   7.6246189861593985f  // ln(2048)
#define LDSPAD    72            // 64 + 8 bf16 pad to break bank conflicts

__device__ __forceinline__ v8f wmma_bf16(v16bf a, v16bf b, v8f c) {
    return __builtin_amdgcn_wmma_f32_16x16x32_bf16(false, a, false, b,
                                                   (short)0, c, false, false);
}

// ---------------------------------------------------------------------------
// Kernel 1: streaming KL + lse pass over the full dense score matrix.
// Block = 256 threads (8 waves), each wave owns a 16-query WMMA tile
// (block covers 128 queries). Keys streamed in 64-key chunks through LDS.
// Per row keep online (m, Z=sum e^{s-m}, T=sum s*e^{s-m}); at the end
// lse = m + log Z,  KL = ln(L) + T/Z - lse.
// ---------------------------------------------------------------------------
#define K1_WAVES  8
#define K1_QROWS  (K1_WAVES * 16)   // 128
#define K1_KCHUNK 64

__global__ __launch_bounds__(256)
void kl_pass_kernel(const float* __restrict__ Q, const float* __restrict__ K,
                    float* __restrict__ kl_out, float* __restrict__ lse_out) {
    const int bh   = blockIdx.x;            // b*16 + h
    const int b    = bh >> 4, h = bh & 15;
    const int q0   = blockIdx.y * K1_QROWS;
    const int tid  = threadIdx.x;
    const int lane = tid & 31, wave = tid >> 5;
    const int half = lane >> 4, mrow = lane & 15;

    __shared__ __bf16 Qs[K1_QROWS][LDSPAD];
    __shared__ __bf16 Ks[K1_KCHUNK][LDSPAD];

    const float* Qbase = Q + ((size_t)b * SEQ) * D_MODEL + (size_t)h * HEAD_DIM;
    const float* Kbase = K + ((size_t)b * SEQ) * D_MODEL + (size_t)h * HEAD_DIM;

    // Stage Q tile (128 rows x 64), fp32 -> bf16
    for (int e = tid; e < K1_QROWS * HEAD_DIM; e += 256) {
        int r = e >> 6, c = e & 63;
        Qs[r][c] = (__bf16)Qbase[(size_t)(q0 + r) * D_MODEL + c];
    }
    __syncthreads();

    // Loop-invariant A fragments (ISA 16-bit A 16x32 layout)
    v16bf a0, a1;
    {
        const int qr = wave * 16 + mrow;
#pragma unroll
        for (int e = 0; e < 8; ++e) {
            a0[e]     = Qs[qr][half * 8 + e];
            a0[e + 8] = Qs[qr][16 + half * 8 + e];
            a1[e]     = Qs[qr][32 + half * 8 + e];
            a1[e + 8] = Qs[qr][48 + half * 8 + e];
        }
    }

    float mrun[8], Zrun[8], Trun[8];
#pragma unroll
    for (int r = 0; r < 8; ++r) { mrun[r] = -1e30f; Zrun[r] = 0.f; Trun[r] = 0.f; }

    for (int kc = 0; kc < SEQ; kc += K1_KCHUNK) {
        __syncthreads();   // previous-iteration readers done before overwrite
        for (int e = tid; e < K1_KCHUNK * HEAD_DIM; e += 256) {
            int r = e >> 6, c = e & 63;
            Ks[r][c] = (__bf16)Kbase[(size_t)(kc + r) * D_MODEL + c];
        }
        if (kc + K1_KCHUNK < SEQ)   // prefetch next key chunk (global_prefetch_b8)
            __builtin_prefetch(&Kbase[(size_t)(kc + K1_KCHUNK + (tid >> 2)) * D_MODEL +
                                      ((tid & 3) << 4)], 0, 1);
        __syncthreads();

        v8f cacc[4];
#pragma unroll
        for (int sub = 0; sub < 4; ++sub) {
            v16bf b0, b1;
            const int kr = sub * 16 + mrow;   // key row == lane's N
#pragma unroll
            for (int e = 0; e < 8; ++e) {
                b0[e]     = Ks[kr][half * 8 + e];
                b0[e + 8] = Ks[kr][16 + half * 8 + e];
                b1[e]     = Ks[kr][32 + half * 8 + e];
                b1[e + 8] = Ks[kr][48 + half * 8 + e];
            }
            v8f acc = {};
            acc = wmma_bf16(a0, b0, acc);
            acc = wmma_bf16(a1, b1, acc);
            cacc[sub] = acc;
        }

        // Online softmax-stat update (reductions over the 16 lanes of a half)
#pragma unroll
        for (int r = 0; r < 8; ++r) {
            float s0 = cacc[0][r] * SCALE, s1 = cacc[1][r] * SCALE;
            float s2 = cacc[2][r] * SCALE, s3 = cacc[3][r] * SCALE;
            float cm = fmaxf(fmaxf(s0, s1), fmaxf(s2, s3));
#pragma unroll
            for (int off = 8; off >= 1; off >>= 1)
                cm = fmaxf(cm, __shfl_xor(cm, off, 32));
            float mn = fmaxf(mrun[r], cm);
            float e0 = __expf(s0 - mn), e1 = __expf(s1 - mn);
            float e2 = __expf(s2 - mn), e3 = __expf(s3 - mn);
            float se = e0 + e1 + e2 + e3;
            float ss = e0 * s0 + e1 * s1 + e2 * s2 + e3 * s3;
#pragma unroll
            for (int off = 8; off >= 1; off >>= 1) {
                se += __shfl_xor(se, off, 32);
                ss += __shfl_xor(ss, off, 32);
            }
            float alpha = __expf(mrun[r] - mn);
            Zrun[r] = Zrun[r] * alpha + se;
            Trun[r] = Trun[r] * alpha + ss;
            mrun[r] = mn;
        }
    }

    if (mrow == 0) {   // lanes 0 and 16 each own 8 rows (C layout: M = r + half*8)
#pragma unroll
        for (int r = 0; r < 8; ++r) {
            int q = q0 + wave * 16 + half * 8 + r;
            float lse = mrun[r] + __logf(Zrun[r]);
            float kl  = LOG_SEQ + Trun[r] / Zrun[r] - lse;
            lse_out[bh * SEQ + q] = lse;
            kl_out[bh * SEQ + q]  = kl;
        }
    }
}

// ---------------------------------------------------------------------------
// Kernel 2: top-38 selection per (b,h), iterative argmax, low-index tie-break
// (matches jax.lax.top_k ordering). Mutates kl in-place (recomputed each call).
// ---------------------------------------------------------------------------
__global__ __launch_bounds__(256)
void topk_kernel(float* __restrict__ kl, int* __restrict__ idx_out) {
    const int bh = blockIdx.x;
    float* klrow = kl + (size_t)bh * SEQ;
    __shared__ float sv[256];
    __shared__ int   si[256];
    const int tid = threadIdx.x;
    for (int it = 0; it < TOP_U; ++it) {
        float bv = -1e30f; int bi = SEQ;
        for (int j = tid; j < SEQ; j += 256) {
            float v = klrow[j];
            if (v > bv || (v == bv && j < bi)) { bv = v; bi = j; }
        }
        sv[tid] = bv; si[tid] = bi;
        __syncthreads();
        for (int s = 128; s > 0; s >>= 1) {
            if (tid < s) {
                float ov = sv[tid + s]; int oi = si[tid + s];
                if (ov > sv[tid] || (ov == sv[tid] && oi < si[tid])) {
                    sv[tid] = ov; si[tid] = oi;
                }
            }
            __syncthreads();
        }
        if (tid == 0) { idx_out[bh * TOP_U + it] = si[0]; klrow[si[0]] = -1e30f; }
        __syncthreads();
    }
}

// ---------------------------------------------------------------------------
// Kernel 3: sparse attention on the 38 selected queries per (b,h).
// p = exp(scale*s - lse) uses the exact lse from kernel 1 -> single pass.
// Waves 0..2 own 16-query tiles (rows 38..47 padded). WMMA for Q·Kᵀ and P·V.
// Writes attn_weights and scatters output rows into the zeroed output.
// ---------------------------------------------------------------------------
#define K3_KCHUNK 32

__global__ __launch_bounds__(128)
void sparse_attn_kernel(const float* __restrict__ Q, const float* __restrict__ K,
                        const float* __restrict__ V,
                        const float* __restrict__ lse_in, const int* __restrict__ idx_in,
                        float* __restrict__ out, float* __restrict__ attnw) {
    const int bh   = blockIdx.x;
    const int b    = bh >> 4, h = bh & 15;
    const int tid  = threadIdx.x;
    const int lane = tid & 31, wave = tid >> 5;
    const int half = lane >> 4, mrow = lane & 15;

    __shared__ __bf16 Qs[48][LDSPAD];
    __shared__ __bf16 Ks[K3_KCHUNK][LDSPAD];
    __shared__ __bf16 Vs[K3_KCHUNK][LDSPAD];
    __shared__ __bf16 Pb[3][16][40];     // per-wave P tile, padded
    __shared__ float  lse_s[48];
    __shared__ int    idx_s[48];

    const float* Qbase = Q + ((size_t)b * SEQ) * D_MODEL + (size_t)h * HEAD_DIM;
    const float* Kbase = K + ((size_t)b * SEQ) * D_MODEL + (size_t)h * HEAD_DIM;
    const float* Vbase = V + ((size_t)b * SEQ) * D_MODEL + (size_t)h * HEAD_DIM;

    if (tid < 48) {
        int qi = (tid < TOP_U) ? idx_in[bh * TOP_U + tid] : 0;
        idx_s[tid] = qi;
        lse_s[tid] = (tid < TOP_U) ? lse_in[bh * SEQ + qi] : 0.f;
    }
    __syncthreads();

    // Gather selected Q rows (padded rows zero)
    for (int e = tid; e < 48 * HEAD_DIM; e += 128) {
        int r = e >> 6, c = e & 63;
        float v = (r < TOP_U) ? Qbase[(size_t)idx_s[r] * D_MODEL + c] : 0.f;
        Qs[r][c] = (__bf16)v;
    }
    __syncthreads();

    v16bf a0 = {}, a1 = {};
    float lse_r[8] = {0.f, 0.f, 0.f, 0.f, 0.f, 0.f, 0.f, 0.f};
    if (wave < 3) {
        const int qr = wave * 16 + mrow;
#pragma unroll
        for (int e = 0; e < 8; ++e) {
            a0[e]     = Qs[qr][half * 8 + e];
            a0[e + 8] = Qs[qr][16 + half * 8 + e];
            a1[e]     = Qs[qr][32 + half * 8 + e];
            a1[e + 8] = Qs[qr][48 + half * 8 + e];
        }
#pragma unroll
        for (int r = 0; r < 8; ++r) lse_r[r] = lse_s[wave * 16 + half * 8 + r];
    }
    v8f o0 = {}, o1 = {}, o2 = {}, o3 = {};
    float* attnbase = attnw + (size_t)bh * TOP_U * SEQ;

    for (int kc = 0; kc < SEQ; kc += K3_KCHUNK) {
        for (int e = tid; e < K3_KCHUNK * HEAD_DIM; e += 128) {
            int r = e >> 6, c = e & 63;
            Ks[r][c] = (__bf16)Kbase[(size_t)(kc + r) * D_MODEL + c];
            Vs[r][c] = (__bf16)Vbase[(size_t)(kc + r) * D_MODEL + c];
        }
        __syncthreads();

        if (wave < 3) {
#pragma unroll
            for (int sub = 0; sub < 2; ++sub) {
                v16bf b0, b1;
                const int kr = sub * 16 + mrow;
#pragma unroll
                for (int e = 0; e < 8; ++e) {
                    b0[e]     = Ks[kr][half * 8 + e];
                    b0[e + 8] = Ks[kr][16 + half * 8 + e];
                    b1[e]     = Ks[kr][32 + half * 8 + e];
                    b1[e + 8] = Ks[kr][48 + half * 8 + e];
                }
                v8f acc = {};
                acc = wmma_bf16(a0, b0, acc);
                acc = wmma_bf16(a1, b1, acc);
#pragma unroll
                for (int r = 0; r < 8; ++r) {
                    const int m    = half * 8 + r;        // C layout row
                    const int urow = wave * 16 + m;
                    float p = __expf(acc[r] * SCALE - lse_r[r]);
                    if (urow < TOP_U)
                        attnbase[(size_t)urow * SEQ + (kc + sub * 16 + mrow)] = p;
                    Pb[wave][m][sub * 16 + mrow] = (__bf16)p;
                }
            }
        }
        __syncthreads();

        if (wave < 3) {
            v16bf pf;
#pragma unroll
            for (int e = 0; e < 8; ++e) {
                pf[e]     = Pb[wave][mrow][half * 8 + e];
                pf[e + 8] = Pb[wave][mrow][16 + half * 8 + e];
            }
#pragma unroll
            for (int dsub = 0; dsub < 4; ++dsub) {
                v16bf vf;
#pragma unroll
                for (int e = 0; e < 8; ++e) {
                    vf[e]     = Vs[half * 8 + e][dsub * 16 + mrow];
                    vf[e + 8] = Vs[16 + half * 8 + e][dsub * 16 + mrow];
                }
                if (dsub == 0) o0 = wmma_bf16(pf, vf, o0);
                else if (dsub == 1) o1 = wmma_bf16(pf, vf, o1);
                else if (dsub == 2) o2 = wmma_bf16(pf, vf, o2);
                else o3 = wmma_bf16(pf, vf, o3);
            }
        }
        __syncthreads();
    }

    if (wave < 3) {
#pragma unroll
        for (int r = 0; r < 8; ++r) {
            const int m    = half * 8 + r;
            const int urow = wave * 16 + m;
            if (urow < TOP_U) {
                const int q = idx_s[urow];
                size_t ob = ((size_t)b * SEQ + q) * D_MODEL + h * HEAD_DIM;
                out[ob +  0 + mrow] = o0[r];
                out[ob + 16 + mrow] = o1[r];
                out[ob + 32 + mrow] = o2[r];
                out[ob + 48 + mrow] = o3[r];
            }
        }
    }
}

// ---------------------------------------------------------------------------
extern "C" void kernel_launch(void* const* d_in, const int* in_sizes, int n_in,
                              void* d_out, int out_size, void* d_ws, size_t ws_size,
                              hipStream_t stream) {
    const float* Q = (const float*)d_in[0];
    const float* K = (const float*)d_in[1];
    const float* V = (const float*)d_in[2];

    float* out   = (float*)d_out;
    float* attnw = out + (size_t)B_SZ * SEQ * D_MODEL;   // second tuple output

    float* kl_ws  = (float*)d_ws;                         // B*H*L floats
    float* lse_ws = kl_ws + (size_t)B_SZ * NHEAD * SEQ;   // B*H*L floats
    int*   idx_ws = (int*)(lse_ws + (size_t)B_SZ * NHEAD * SEQ);  // B*H*U ints

    // Non-selected output rows must be zero every call (graph-capturable).
    hipMemsetAsync(d_out, 0, (size_t)B_SZ * SEQ * D_MODEL * sizeof(float), stream);

    dim3 g1(B_SZ * NHEAD, SEQ / K1_QROWS);
    kl_pass_kernel<<<g1, 256, 0, stream>>>(Q, K, kl_ws, lse_ws);
    topk_kernel<<<B_SZ * NHEAD, 256, 0, stream>>>(kl_ws, idx_ws);
    sparse_attn_kernel<<<B_SZ * NHEAD, 128, 0, stream>>>(Q, K, V, lse_ws, idx_ws,
                                                         out, attnw);
}
